// SpaceTransformer_elsa_63814624084312
// MI455X (gfx1250) — compile-verified, matrix operand order checked by reading, not run on hardware
//
#include <hip/hip_runtime.h>
#include <hip/hip_bf16.h>

typedef _Float16 half_t;
typedef __attribute__((ext_vector_type(16))) _Float16 v16h;
typedef __attribute__((ext_vector_type(8)))  _Float16 v8h;
typedef __attribute__((ext_vector_type(8)))  float    v8f;

#define NT_TOK   16384      // Bp*H*W = 4*64*64
#define CDIM     192
#define QKVN     576
#define NHD      6
#define K2D      49
#define ATTN_LD  294        // NH*K2
#define HIDN     384

// x / out layout index: (B,C,D,H,W) with B=2,C=192,D=2,H=64,W=64
__device__ __forceinline__ size_t xidx(int m, int c) {
    int bp  = m >> 12;          // token -> (b,d)
    int b   = bp >> 1;
    int d   = bp & 1;
    int rem = m & 4095;         // h*64+w
    return (size_t)b * 1572864u + (size_t)c * 8192u + (size_t)d * 4096u + (size_t)rem;
}

// ---------------------------------------------------------------------------
// Repack a KxN f32 weight matrix into WMMA B-fragment order (f16), tiles of
// 32x16, 512 halves per tile, 16 contiguous halves per lane. Zero-pads both
// K (to a multiple of 32) and N (to a multiple of 64 = 4 tiles).
// ---------------------------------------------------------------------------
__global__ void repack_b(const float* __restrict__ W, int Ksrc, int Nsrc,
                         int Ntiles, half_t* __restrict__ out, int total) {
    int i = blockIdx.x * 256 + threadIdx.x;
    if (i >= total) return;
    int j    = i & 15;          // half index within lane
    int l    = (i >> 4) & 31;   // lane
    int tile = i >> 9;
    int nt   = tile % Ntiles;
    int kt   = tile / Ntiles;
    int g    = l >> 4;
    int n    = nt * 16 + (l & 15);
    int k    = kt * 32 + ((j < 8) ? (g * 8 + j) : (16 + g * 8 + (j - 8)));
    float v  = (k < Ksrc && n < Nsrc) ? W[(size_t)k * Nsrc + n] : 0.0f;
    out[i] = (half_t)v;
}

// ---------------------------------------------------------------------------
// LayerNorm over C=192, reading x in (B,C,D,H,W) layout, writing f16 rows.
// ---------------------------------------------------------------------------
__global__ void ln_from_x(const float* __restrict__ x, const float* __restrict__ w,
                          const float* __restrict__ bb, half_t* __restrict__ y) {
    int t = blockIdx.x * blockDim.x + threadIdx.x;
    if (t >= NT_TOK) return;
    int bp = t >> 12, b = bp >> 1, d = bp & 1, rem = t & 4095;
    size_t base = (size_t)b * 1572864u + (size_t)d * 4096u + (size_t)rem;
    float s = 0.f, s2 = 0.f;
    for (int c = 0; c < CDIM; ++c) {
        float v = x[base + (size_t)c * 8192u];
        s += v; s2 += v * v;
    }
    float mean = s * (1.0f / CDIM);
    float var  = s2 * (1.0f / CDIM) - mean * mean;
    float rr   = rsqrtf(var + 1e-5f);
    for (int c = 0; c < CDIM; ++c) {
        float v = (x[base + (size_t)c * 8192u] - mean) * rr * w[c] + bb[c];
        y[(size_t)t * CDIM + c] = (half_t)v;
    }
}

__global__ void ln_rowmajor(const float* __restrict__ in, const float* __restrict__ w,
                            const float* __restrict__ bb, half_t* __restrict__ y) {
    int t = blockIdx.x * blockDim.x + threadIdx.x;
    if (t >= NT_TOK) return;
    const float* r = in + (size_t)t * CDIM;
    float s = 0.f, s2 = 0.f;
    for (int c = 0; c < CDIM; ++c) { float v = r[c]; s += v; s2 += v * v; }
    float mean = s * (1.0f / CDIM);
    float var  = s2 * (1.0f / CDIM) - mean * mean;
    float rr   = rsqrtf(var + 1e-5f);
    for (int c = 0; c < CDIM; ++c) {
        float v = (r[c] - mean) * rr * w[c] + bb[c];
        y[(size_t)t * CDIM + c] = (half_t)v;
    }
}

// ---------------------------------------------------------------------------
// WMMA GEMM: one wave computes a 16x64 C strip (4 N-tiles) so the A fragment
// is reused across 4 v_wmma issues. Ntiles is padded to a multiple of 4, so
// the K-loop is completely branch-free (EXEC all-ones throughout).
// Epilogue modes: 0=none, 1=exact GELU, 2=per-column scale/shift.
// ---------------------------------------------------------------------------
__global__ void gemm16x4_wmma(const half_t* __restrict__ A, int lda,
                              const half_t* __restrict__ Bp, int Ntiles, // padded %4==0
                              const float* __restrict__ bias, int mode,
                              const float* __restrict__ colscale,
                              const float* __restrict__ colshift,
                              const float* __restrict__ resid, int resid_is_x,
                              float* __restrict__ outF, half_t* __restrict__ outH,
                              float* __restrict__ outX,
                              int ldc, int Ncols, int K) {
    int wave = threadIdx.x >> 5;
    int lane = threadIdx.x & 31;
    int nt0  = (blockIdx.y * 4 + wave) * 4;     // first N-tile of this wave's strip
    if (nt0 >= Ntiles) return;                  // wave-uniform
    int mt = blockIdx.x;
    int g  = lane >> 4;
    int ml = lane & 15;

    const half_t* arow = A + (size_t)(mt * 16 + ml) * lda + g * 8;
    v8f acc0 = {}, acc1 = {}, acc2 = {}, acc3 = {};

    int kt_n = K >> 5;
    for (int kt = 0; kt < kt_n; ++kt) {
        v16h a;
        v8h* ap = (v8h*)&a;
        ap[0] = *(const v8h*)(arow + kt * 32);
        ap[1] = *(const v8h*)(arow + kt * 32 + 16);

        const half_t* bbase = Bp + ((size_t)kt * Ntiles + nt0) * 512 + (size_t)lane * 16;
        v16h b0, b1, b2, b3;
        ((v8h*)&b0)[0] = *(const v8h*)(bbase);
        ((v8h*)&b0)[1] = *(const v8h*)(bbase + 8);
        ((v8h*)&b1)[0] = *(const v8h*)(bbase + 512);
        ((v8h*)&b1)[1] = *(const v8h*)(bbase + 512 + 8);
        ((v8h*)&b2)[0] = *(const v8h*)(bbase + 1024);
        ((v8h*)&b2)[1] = *(const v8h*)(bbase + 1024 + 8);
        ((v8h*)&b3)[0] = *(const v8h*)(bbase + 1536);
        ((v8h*)&b3)[1] = *(const v8h*)(bbase + 1536 + 8);
        acc0 = __builtin_amdgcn_wmma_f32_16x16x32_f16(false, a, false, b0, (short)0, acc0, false, false);
        acc1 = __builtin_amdgcn_wmma_f32_16x16x32_f16(false, a, false, b1, (short)0, acc1, false, false);
        acc2 = __builtin_amdgcn_wmma_f32_16x16x32_f16(false, a, false, b2, (short)0, acc2, false, false);
        acc3 = __builtin_amdgcn_wmma_f32_16x16x32_f16(false, a, false, b3, (short)0, acc3, false, false);
    }

    v8f* accs[4] = { &acc0, &acc1, &acc2, &acc3 };
#pragma unroll
    for (int u = 0; u < 4; ++u) {
        int n = (nt0 + u) * 16 + ml;
        if (n >= Ncols) continue;               // padded columns: skip store
        float bs = bias ? bias[n] : 0.0f;
        float cs = 1.0f, cf = 0.0f;
        if (mode == 2) { cs = colscale[n]; cf = colshift[n]; }
        v8f acc = *accs[u];
#pragma unroll
        for (int r = 0; r < 8; ++r) {
            int m   = mt * 16 + r + 8 * g;
            float v = acc[r] + bs;
            if (mode == 1)      v = 0.5f * v * (1.0f + erff(v * 0.70710678118654752f));
            else if (mode == 2) v = v * cs + cf;
            if (resid) v += resid_is_x ? resid[xidx(m, n)] : resid[(size_t)m * ldc + n];
            if (outX) outX[xidx(m, n)] = v;
            if (outF) outF[(size_t)m * ldc + n] = v;
            if (outH) outH[(size_t)m * ldc + n] = (half_t)v;
        }
    }
}

// qk = q * k * SCALE  (SCALE = 32^-0.5), f16 output
__global__ void make_qk(const float* __restrict__ qkv, half_t* __restrict__ qk, int total) {
    int i = blockIdx.x * 256 + threadIdx.x;
    if (i >= total) return;
    int t = i / CDIM, c = i - t * CDIM;
    const float* r = qkv + (size_t)t * QKVN;
    qk[i] = (half_t)(r[c] * r[c + CDIM] * 0.17677669529663687f);
}

// softmax over K2=49 per (token, head); rows contiguous in attn buffer
__global__ void softmax49(float* __restrict__ attn, int nrows) {
    int i = blockIdx.x * 256 + threadIdx.x;
    if (i >= nrows) return;
    float* p = attn + (size_t)i * K2D;   // row = token*6 + head, stride 49
    float mx = -1e30f;
    for (int k = 0; k < K2D; ++k) mx = fmaxf(mx, p[k]);
    float s = 0.f;
    for (int k = 0; k < K2D; ++k) { float e = __expf(p[k] - mx); p[k] = e; s += e; }
    float rr = 1.0f / s;
    for (int k = 0; k < K2D; ++k) p[k] *= rr;
}

// agg[t, c] = sum_k attn[t, n*49+k] * v[neighbor_k(t), c]   (c = n*32+hd)
__global__ void aggregate(const float* __restrict__ attn, const float* __restrict__ qkv,
                          half_t* __restrict__ agg) {
    int t = blockIdx.x;
    int c = threadIdx.x;                 // 0..191
    __shared__ float sat[ATTN_LD];
    for (int i = c; i < ATTN_LD; i += CDIM) sat[i] = attn[(size_t)t * ATTN_LD + i];
    __syncthreads();
    int n  = c >> 5;
    int bp = t >> 12, rem = t & 4095, h = rem >> 6, w = rem & 63;
    float sum = 0.f;
    for (int ky = 0; ky < 7; ++ky) {
        int hy = h + ky - 3;
        if (hy < 0 || hy > 63) continue;
        for (int kx = 0; kx < 7; ++kx) {
            int wx = w + kx - 3;
            if (wx < 0 || wx > 63) continue;
            int tt = (bp << 12) + (hy << 6) + wx;
            sum += sat[n * K2D + ky * 7 + kx] * qkv[(size_t)tt * QKVN + 384 + c];
        }
    }
    agg[(size_t)t * CDIM + c] = (half_t)sum;
}

// ---------------------------------------------------------------------------
// Workspace layout (bytes)
// ---------------------------------------------------------------------------
#define OFF_YH    ((size_t)0)            // 16384*192 f16
#define OFF_QKV   ((size_t)6291456)      // 16384*576 f32
#define OFF_QKH   ((size_t)44040192)     // 16384*192 f16
#define OFF_HH    ((size_t)50331648)     // 16384*64  f16 (K-padded to 64)
#define OFF_ATTN  ((size_t)52428800)     // 16384*294 f32
#define OFF_AGG   ((size_t)71696384)     // 16384*192 f16
#define OFF_H1    ((size_t)77987840)     // 16384*192 f32
#define OFF_ZH    ((size_t)90570752)     // 16384*192 f16
#define OFF_TH    ((size_t)96862208)     // 16384*384 f16
#define OFF_WPK   ((size_t)109445120)    // packed weights (N-tiles padded to %4)
#define PK_PRE    (OFF_WPK + 0)          // 6*36*512 halves
#define PK_A1     (OFF_WPK + 221184)     // 6*4*512
#define PK_A2     (OFF_WPK + 245760)     // 2*20*512
#define PK_PROJ   (OFF_WPK + 286720)     // 6*12*512
#define PK_FC1    (OFF_WPK + 360448)     // 6*24*512
#define PK_FC2    (OFF_WPK + 507904)     // 12*12*512

extern "C" void kernel_launch(void* const* d_in, const int* in_sizes, int n_in,
                              void* d_out, int out_size, void* d_ws, size_t ws_size,
                              hipStream_t stream) {
    const float* x      = (const float*)d_in[0];
    const float* n1_w   = (const float*)d_in[1];
    const float* n1_b   = (const float*)d_in[2];
    const float* pre_w  = (const float*)d_in[3];
    const float* pre_b  = (const float*)d_in[4];
    const float* a1_w   = (const float*)d_in[5];
    const float* a1_b   = (const float*)d_in[6];
    const float* a2_w   = (const float*)d_in[7];
    const float* a2_b   = (const float*)d_in[8];
    const float* gh     = (const float*)d_in[9];   // (2,6,49) -> gh0 = gh, gh1 = gh+294
    const float* proj_w = (const float*)d_in[10];
    const float* proj_b = (const float*)d_in[11];
    const float* n2_w   = (const float*)d_in[12];
    const float* n2_b   = (const float*)d_in[13];
    const float* fc1_w  = (const float*)d_in[14];
    const float* fc1_b  = (const float*)d_in[15];
    const float* fc2_w  = (const float*)d_in[16];
    const float* fc2_b  = (const float*)d_in[17];
    float* out = (float*)d_out;

    char* ws = (char*)d_ws;
    half_t* y_h   = (half_t*)(ws + OFF_YH);
    float*  qkv   = (float*) (ws + OFF_QKV);
    half_t* qk_h  = (half_t*)(ws + OFF_QKH);
    half_t* hh_h  = (half_t*)(ws + OFF_HH);
    float*  attn  = (float*) (ws + OFF_ATTN);
    half_t* agg_h = (half_t*)(ws + OFF_AGG);
    float*  h1    = (float*) (ws + OFF_H1);
    half_t* z_h   = (half_t*)(ws + OFF_ZH);
    half_t* t_h   = (half_t*)(ws + OFF_TH);
    half_t* pk_pre  = (half_t*)(ws + PK_PRE);
    half_t* pk_a1   = (half_t*)(ws + PK_A1);
    half_t* pk_a2   = (half_t*)(ws + PK_A2);
    half_t* pk_proj = (half_t*)(ws + PK_PROJ);
    half_t* pk_fc1  = (half_t*)(ws + PK_FC1);
    half_t* pk_fc2  = (half_t*)(ws + PK_FC2);

    // 1) repack weights into WMMA fragment order (f16, K- and N-padded)
    struct { const float* w; int Ks, Ns, Nt, total; half_t* dst; } rp[6] = {
        { pre_w,  192, 576, 36, 6  * 36 * 512, pk_pre  },
        { a1_w,   192,  48,  4, 6  *  4 * 512, pk_a1   },   // N padded 3->4 tiles
        { a2_w,    48, 294, 20, 2  * 20 * 512, pk_a2   },   // K 48->64, N 19->20 tiles
        { proj_w, 192, 192, 12, 6  * 12 * 512, pk_proj },
        { fc1_w,  192, 384, 24, 6  * 24 * 512, pk_fc1  },
        { fc2_w,  384, 192, 12, 12 * 12 * 512, pk_fc2  },
    };
    for (int i = 0; i < 6; ++i)
        repack_b<<<(rp[i].total + 255) / 256, 256, 0, stream>>>(
            rp[i].w, rp[i].Ks, rp[i].Ns, rp[i].Nt, rp[i].dst, rp[i].total);

    // 2) LN1 -> f16
    ln_from_x<<<NT_TOK / 256, 256, 0, stream>>>(x, n1_w, n1_b, y_h);

    // 3) qkv = y @ pre_w + pre_b   (36 tiles -> 9 wave-strips)
    gemm16x4_wmma<<<dim3(1024, 3), 128, 0, stream>>>(
        y_h, 192, pk_pre, 36, pre_b, 0, nullptr, nullptr,
        nullptr, 0, qkv, nullptr, nullptr, QKVN, QKVN, 192);

    // 4) qk = q*k*scale -> f16
    make_qk<<<(NT_TOK * CDIM) / 256, 256, 0, stream>>>(qkv, qk_h, NT_TOK * CDIM);

    // 5) hh = gelu(qk @ a1_w + a1_b)  (4 padded tiles -> 1 strip)
    hipMemsetAsync(hh_h, 0, (size_t)NT_TOK * 64 * sizeof(half_t), stream);
    gemm16x4_wmma<<<dim3(1024, 1), 128, 0, stream>>>(
        qk_h, 192, pk_a1, 4, a1_b, 1, nullptr, nullptr,
        nullptr, 0, nullptr, hh_h, nullptr, 64, 48, 192);

    // 6) logits = (hh @ a2_w + a2_b) * gh0 + gh1   (20 padded tiles -> 5 strips)
    gemm16x4_wmma<<<dim3(1024, 2), 128, 0, stream>>>(
        hh_h, 64, pk_a2, 20, a2_b, 2, gh, gh + ATTN_LD,
        nullptr, 0, attn, nullptr, nullptr, ATTN_LD, ATTN_LD, 64);

    // 7) softmax over 49 per (token, head)
    softmax49<<<(NT_TOK * NHD) / 256, 256, 0, stream>>>(attn, NT_TOK * NHD);

    // 8) neighborhood aggregation -> agg (f16)
    aggregate<<<NT_TOK, CDIM, 0, stream>>>(attn, qkv, agg_h);

    // 9) h1 = h0 + agg @ proj_w + proj_b  (12 tiles -> 3 strips)
    gemm16x4_wmma<<<dim3(1024, 1), 128, 0, stream>>>(
        agg_h, 192, pk_proj, 12, proj_b, 0, nullptr, nullptr,
        x, 1, h1, nullptr, nullptr, CDIM, CDIM, 192);

    // 10) z = LN2(h1) -> f16
    ln_rowmajor<<<NT_TOK / 256, 256, 0, stream>>>(h1, n2_w, n2_b, z_h);

    // 11) t = gelu(z @ fc1_w + fc1_b)  (24 tiles -> 6 strips)
    gemm16x4_wmma<<<dim3(1024, 2), 128, 0, stream>>>(
        z_h, 192, pk_fc1, 24, fc1_b, 1, nullptr, nullptr,
        nullptr, 0, nullptr, t_h, nullptr, HIDN, HIDN, 192);

    // 12) out = h1 + t @ fc2_w + fc2_b, scattered to (B,C,D,H,W)  (12 tiles -> 3 strips)
    gemm16x4_wmma<<<dim3(1024, 1), 128, 0, stream>>>(
        t_h, HIDN, pk_fc2, 12, fc2_b, 0, nullptr, nullptr,
        h1, 0, nullptr, nullptr, out, CDIM, CDIM, HIDN);

    (void)in_sizes; (void)n_in; (void)out_size; (void)ws_size;
}
